// DSP_1872605741348
// MI455X (gfx1250) — matrix operand order repaired and checked by reading, not compiled, not run on hardware
//
#include <hip/hip_runtime.h>
#include <hip/hip_bf16.h>

typedef __attribute__((ext_vector_type(16))) _Float16 v16h;
typedef __attribute__((ext_vector_type(8)))  _Float16 v8h;
typedef __attribute__((ext_vector_type(4)))  _Float16 v4h;
typedef __attribute__((ext_vector_type(2)))  _Float16 v2h;
typedef __attribute__((ext_vector_type(8)))  float    v8f;

#define CCH   64
#define FIN   13
#define TRI   91      // 13*14/2 upper-triangular Gram entries
#define TPB   128     // 4 wave32 per block in fused kernels
#define LSTR  136     // halfs per LDS row (128 + 8 pad -> bank-conflict-free b128 reads)
#define BNEPS 1e-3f

// ---------------------------------------------------------------- zero
__global__ void k_zero(float* __restrict__ p, long n) {
  long i = (long)blockIdx.x * blockDim.x + threadIdx.x;
  long stride = (long)gridDim.x * blockDim.x;
  for (; i < n; i += stride) p[i] = 0.0f;
}

// ------------------------------------------- BN0 stats via Gram matrix
// Accumulates sum(feat) [13] and upper-tri of sum(feat feat^T) [91].
__global__ void k_bn0_stats(const float* __restrict__ points,
                            const float* __restrict__ fc,
                            const float* __restrict__ fcl,
                            const float* __restrict__ fr,
                            float* __restrict__ sOut, int n) {
  float tri[TRI], sm[FIN];
  #pragma unroll
  for (int t = 0; t < TRI; ++t) tri[t] = 0.f;
  #pragma unroll
  for (int k = 0; k < FIN; ++k) sm[k] = 0.f;
  int stride = gridDim.x * blockDim.x;
  for (int i = blockIdx.x * blockDim.x + threadIdx.x; i < n; i += stride) {
    float f[FIN];
    f[0]=fc[i*3+0];  f[1]=fc[i*3+1];  f[2]=fc[i*3+2];
    f[3]=points[i*5+1]; f[4]=points[i*5+2]; f[5]=points[i*5+3]; f[6]=points[i*5+4];
    f[7]=fcl[i*3+0]; f[8]=fcl[i*3+1]; f[9]=fcl[i*3+2];
    f[10]=fr[i*3+0]; f[11]=fr[i*3+1]; f[12]=fr[i*3+2];
    int t = 0;
    #pragma unroll
    for (int a = 0; a < FIN; ++a) {
      sm[a] += f[a];
      #pragma unroll
      for (int b = a; b < FIN; ++b) { tri[t] += f[a] * f[b]; ++t; }
    }
  }
  __shared__ float red[TRI + FIN];
  for (int t = threadIdx.x; t < TRI + FIN; t += blockDim.x) red[t] = 0.f;
  __syncthreads();
  #pragma unroll
  for (int t = 0; t < TRI; ++t) atomicAdd(&red[t], tri[t]);
  #pragma unroll
  for (int k = 0; k < FIN; ++k) atomicAdd(&red[TRI + k], sm[k]);
  __syncthreads();
  for (int t = threadIdx.x; t < TRI + FIN; t += blockDim.x)
    atomicAdd(&sOut[t], red[t]);
}

// mean_c = (sfeat . W0col)/N ; E[y^2]_c = (W0col^T S W0col)/N
__global__ void k_bn0_final(const float* __restrict__ W0,
                            const float* __restrict__ g0,
                            const float* __restrict__ be0,
                            const float* __restrict__ ws,
                            float* __restrict__ a0o, float* __restrict__ b0o,
                            float invN) {
  int c = threadIdx.x;
  if (c >= CCH) return;
  float w[FIN];
  #pragma unroll
  for (int k = 0; k < FIN; ++k) w[k] = W0[k * CCH + c];
  float mean = 0.f;
  #pragma unroll
  for (int k = 0; k < FIN; ++k) mean += ws[TRI + k] * w[k];
  mean *= invN;
  float ex2 = 0.f; int t = 0;
  #pragma unroll
  for (int a = 0; a < FIN; ++a) {
    #pragma unroll
    for (int b = a; b < FIN; ++b) {
      ex2 += (a == b ? 1.f : 2.f) * w[a] * w[b] * ws[t]; ++t;
    }
  }
  ex2 *= invN;
  float var = ex2 - mean * mean;
  float s = g0[c] * rsqrtf(var + BNEPS);
  a0o[c] = s;
  b0o[c] = be0[c] - mean * s;
}

__global__ void k_w1_half(const float* __restrict__ W1, _Float16* __restrict__ w1h) {
  int i = blockIdx.x * blockDim.x + threadIdx.x;
  if (i < 2 * CCH * CCH) w1h[i] = (_Float16)W1[i];
}

// ------------------------------------------- scatter-mean accumulation
__global__ void k_scatter(const float* __restrict__ points,
                          const float* __restrict__ fc,
                          const float* __restrict__ fcl,
                          const float* __restrict__ fr,
                          const int* __restrict__ unq,
                          const float* __restrict__ W0,
                          const float* __restrict__ a0,
                          const float* __restrict__ b0,
                          float* __restrict__ seg, float* __restrict__ cnt, int n) {
  int i = blockIdx.x * blockDim.x + threadIdx.x;
  if (i >= n) return;
  float f[FIN];
  f[0]=fc[i*3+0];  f[1]=fc[i*3+1];  f[2]=fc[i*3+2];
  f[3]=points[i*5+1]; f[4]=points[i*5+2]; f[5]=points[i*5+3]; f[6]=points[i*5+4];
  f[7]=fcl[i*3+0]; f[8]=fcl[i*3+1]; f[9]=fcl[i*3+2];
  f[10]=fr[i*3+0]; f[11]=fr[i*3+1]; f[12]=fr[i*3+2];
  int g = unq[i];
  atomicAdd(&cnt[g], 1.0f);
  float* segRow = seg + (long)g * CCH;
  for (int c = 0; c < CCH; ++c) {        // c uniform -> W0/a0/b0 scalar loads
    float acc = 0.f;
    #pragma unroll
    for (int k = 0; k < FIN; ++k) acc += f[k] * W0[k * CCH + c];
    float h = acc * a0[c] + b0[c];
    h = h > 0.f ? h : 0.f;
    atomicAdd(&segRow[c], h);
  }
}

__global__ void k_mean_final(float* __restrict__ seg, const float* __restrict__ cnt, int total) {
  int i = blockIdx.x * blockDim.x + threadIdx.x;
  if (i >= total) return;
  float c = cnt[i >> 6];
  seg[i] = seg[i] / fmaxf(c, 1.0f);
}

__global__ void k_bn1_final(const float* __restrict__ g1, const float* __restrict__ be1,
                            const float* __restrict__ sum, const float* __restrict__ sq,
                            float* __restrict__ a1, float* __restrict__ b1, float invN) {
  int c = threadIdx.x;
  if (c >= CCH) return;
  float mean = sum[c] * invN;
  float var  = sq[c] * invN - mean * mean;
  float s = g1[c] * rsqrtf(var + BNEPS);
  a1[c] = s;
  b1[c] = be1[c] - mean * s;
}

// ------------------------------------------- fused WMMA pass
// FINAL==0: accumulate BN1 sum/sumsq over y1.  FINAL==1: BN1+ReLU+atomic-max.
template <int FINAL>
__global__ __launch_bounds__(TPB) void k_fused(
    const float* __restrict__ points, const float* __restrict__ fc,
    const float* __restrict__ fcl, const float* __restrict__ fr,
    const int* __restrict__ unq,
    const float* __restrict__ W0,
    const float* __restrict__ a0, const float* __restrict__ b0,
    const float* __restrict__ seg_mean,
    const _Float16* __restrict__ w1h,
    float* __restrict__ bn1sum, float* __restrict__ bn1sq,
    const float* __restrict__ a1, const float* __restrict__ b1,
    unsigned int* __restrict__ vox,
    int n, int nChunks) {
  __shared__ __align__(16) _Float16 h2[TPB * LSTR];
  __shared__ int gl[TPB];

  const int tid  = threadIdx.x;
  const int lane = tid & 31;
  const int wv   = tid >> 5;                 // 0..3 == output-channel tile
  const int hi   = (lane < 16) ? 0 : 1;
  const int col  = (wv << 4) + (lane & 15);  // global output channel for C frag

  // B fragments (contiguous k order: lanes 0-15 -> K 0..15, lanes 16-31 -> K 16..31)
  v16h bfrag[4];
  #pragma unroll
  for (int t = 0; t < 4; ++t) {
    int ks = t * 32 + hi * 16;
    #pragma unroll
    for (int j = 0; j < 16; ++j)
      bfrag[t][j] = w1h[(ks + j) * CCH + col];
  }

  float a1c = 0.f, b1c = 0.f;
  if (FINAL) { a1c = a1[col]; b1c = b1[col]; }
  float ssum = 0.f, sqq = 0.f;

  for (int ch = blockIdx.x; ch < nChunks; ch += gridDim.x) {
    int i = ch * TPB + tid;
    _Float16* row = &h2[tid * LSTR];
    if (i < n) {
      float f[FIN];
      f[0]=fc[i*3+0];  f[1]=fc[i*3+1];  f[2]=fc[i*3+2];
      f[3]=points[i*5+1]; f[4]=points[i*5+2]; f[5]=points[i*5+3]; f[6]=points[i*5+4];
      f[7]=fcl[i*3+0]; f[8]=fcl[i*3+1]; f[9]=fcl[i*3+2];
      f[10]=fr[i*3+0]; f[11]=fr[i*3+1]; f[12]=fr[i*3+2];
      int g = unq[i];
      gl[tid] = g;
      for (int c = 0; c < CCH; c += 2) {   // c uniform -> scalar W0/a0/b0 loads
        float acc0 = 0.f, acc1 = 0.f;
        #pragma unroll
        for (int k = 0; k < FIN; ++k) {
          acc0 += f[k] * W0[k * CCH + c];
          acc1 += f[k] * W0[k * CCH + c + 1];
        }
        float h0 = acc0 * a0[c] + b0[c];
        float h1 = acc1 * a0[c + 1] + b0[c + 1];
        v2h p; p[0] = (_Float16)(h0 > 0.f ? h0 : 0.f);
        p[1] = (_Float16)(h1 > 0.f ? h1 : 0.f);
        *(v2h*)&row[c] = p;
      }
      const float4* mrow = (const float4*)(seg_mean + (long)g * CCH);
      #pragma unroll
      for (int q = 0; q < 16; ++q) {
        float4 m = mrow[q];
        v4h mh; mh[0]=(_Float16)m.x; mh[1]=(_Float16)m.y;
        mh[2]=(_Float16)m.z; mh[3]=(_Float16)m.w;
        *(v4h*)&row[CCH + q * 4] = mh;
      }
    } else {
      gl[tid] = -1;
      v8h z;
      #pragma unroll
      for (int j = 0; j < 8; ++j) z[j] = (_Float16)0.f;
      #pragma unroll
      for (int q = 0; q < 16; ++q) *(v8h*)&row[q * 8] = z;   // zero padded row
    }
    __syncthreads();

    // each wave: 8 row-tiles (16 points) x K=128 of its channel tile
    #pragma unroll 1
    for (int r = 0; r < 8; ++r) {
      const _Float16* ap = &h2[(r * 16 + (lane & 15)) * LSTR];
      v8f acc = {};
      #pragma unroll
      for (int t = 0; t < 4; ++t) {
        // A layout matching B: lanes 0-15 hold K 0..15, lanes 16-31 hold K 16..31
        int off = t * 32 + hi * 16;
        v8h p0 = *(const v8h*)&ap[off];
        v8h p1 = *(const v8h*)&ap[off + 8];
        v16h af;
        #pragma unroll
        for (int j = 0; j < 8; ++j) { af[j] = p0[j]; af[j + 8] = p1[j]; }
        acc = __builtin_amdgcn_wmma_f32_16x16x32_f16(
            false, af, false, bfrag[t], (short)0, acc, false, false);
      }
      if (FINAL) {
        #pragma unroll
        for (int v = 0; v < 8; ++v) {
          float val = acc[v] * a1c + b1c;
          val = val > 0.f ? val : 0.f;                 // >= 0 -> uint bits monotonic
          int p = r * 16 + v + hi * 8;                 // C frag row -> point in chunk
          int g = gl[p];
          if (g >= 0)
            atomicMax(&vox[(long)g * CCH + col], __float_as_uint(val));
        }
      } else {
        #pragma unroll
        for (int v = 0; v < 8; ++v) { float y = acc[v]; ssum += y; sqq += y * y; }
      }
    }
    __syncthreads();
  }

  if (!FINAL) {
    atomicAdd(&bn1sum[col], ssum);
    atomicAdd(&bn1sq[col],  sqq);
  }
}

// ---------------------------------------------------------------- launch
extern "C" void kernel_launch(void* const* d_in, const int* in_sizes, int n_in,
                              void* d_out, int out_size, void* d_ws, size_t ws_size,
                              hipStream_t stream) {
  const float* points = (const float*)d_in[0];
  const float* fc     = (const float*)d_in[1];
  const float* fcl    = (const float*)d_in[2];
  const float* fr     = (const float*)d_in[3];
  const int*   unq    = (const int*)  d_in[4];
  const float* W0     = (const float*)d_in[5];
  const float* g0     = (const float*)d_in[6];
  const float* be0    = (const float*)d_in[7];
  const float* W1     = (const float*)d_in[8];
  const float* g1     = (const float*)d_in[9];
  const float* be1    = (const float*)d_in[10];

  const int n = in_sizes[0] / 5;
  const int V = out_size / CCH;

  float* wsf    = (float*)d_ws;
  float* stats  = wsf;              // [0..103] Gram tri + sfeat
  float* bn1sum = wsf + 108;
  float* bn1sq  = wsf + 172;
  float* a0     = wsf + 236;
  float* b0     = wsf + 300;
  float* a1     = wsf + 364;
  float* b1     = wsf + 428;
  _Float16* w1h = (_Float16*)(wsf + 512);          // 8192 halfs
  float* cnt    = wsf + 4608;                      // V floats
  int    vpad   = (V + 3) & ~3;
  float* seg    = wsf + 4608 + vpad;               // V*64 floats (sum -> mean in place)

  long zTotal = 4608 + (long)vpad + (long)V * CCH;
  k_zero<<<1024, 256, 0, stream>>>(wsf, zTotal);
  k_zero<<<1024, 256, 0, stream>>>((float*)d_out, (long)out_size);

  k_bn0_stats<<<512, 256, 0, stream>>>(points, fc, fcl, fr, stats, n);
  k_bn0_final<<<1, 64, 0, stream>>>(W0, g0, be0, stats, a0, b0, 1.0f / (float)n);
  k_w1_half<<<(2 * CCH * CCH + 255) / 256, 256, 0, stream>>>(W1, w1h);

  k_scatter<<<(n + 255) / 256, 256, 0, stream>>>(points, fc, fcl, fr, unq,
                                                 W0, a0, b0, seg, cnt, n);
  k_mean_final<<<(V * CCH + 255) / 256, 256, 0, stream>>>(seg, cnt, V * CCH);

  const int nChunks = (n + TPB - 1) / TPB;
  k_fused<0><<<1024, TPB, 0, stream>>>(points, fc, fcl, fr, unq, W0, a0, b0,
                                       seg, w1h, bn1sum, bn1sq, a1, b1,
                                       (unsigned int*)d_out, n, nChunks);
  k_bn1_final<<<1, 64, 0, stream>>>(g1, be1, bn1sum, bn1sq, a1, b1, 1.0f / (float)n);
  k_fused<1><<<1024, TPB, 0, stream>>>(points, fc, fcl, fr, unq, W0, a0, b0,
                                       seg, w1h, bn1sum, bn1sq, a1, b1,
                                       (unsigned int*)d_out, n, nChunks);
}
